// FastfoodWrapper_2491081032315
// MI455X (gfx1250) — matrix operand order repaired and checked by reading, not compiled, not run on hardware
//
#include <hip/hip_runtime.h>

#define Bsz   1024
#define DIN   2048
#define DH    4096
#define DOUT  2048
#define DINT  1024
#define LLW   (1 << 23)

typedef __attribute__((ext_vector_type(16))) __bf16 v16bf;
typedef __attribute__((ext_vector_type(8)))  float  v8f;

struct alignas(16) U128 { unsigned int x[4]; };
union FragU { U128 u[2]; v16bf v; };

static __device__ inline unsigned short f2bf(float f) {
  union { float f; unsigned u; } a; a.f = f;
  unsigned r = a.u + 0x7FFFu + ((a.u >> 16) & 1u);   // round-to-nearest-even
  return (unsigned short)(r >> 16);
}

// ---------------- fastfood building blocks ----------------

// whA = fwht_1024(BB[:1024] * V)   (first big FWHT collapses to this)
__global__ __launch_bounds__(512) void k_whA(const float* __restrict__ V,
                                             const float* __restrict__ BB,
                                             float* __restrict__ whA) {
  __shared__ float lds[1024];
  const int t = threadIdx.x;
  lds[t]       = BB[t]       * V[t];
  lds[t + 512] = BB[t + 512] * V[t + 512];
  __syncthreads();
  for (int lev = 0; lev < 10; ++lev) {
    const int s = 1 << lev;
    const int i = ((t >> lev) << (lev + 1)) | (t & (s - 1));
    const int j = i + s;
    const float a = lds[i], b = lds[j];
    lds[i] = a + b; lds[j] = a - b;
    __syncthreads();
  }
  whA[t] = lds[t]; whA[t + 512] = lds[t + 512];
}

__global__ __launch_bounds__(256) void k_gg2_partial(const float* __restrict__ g, int n,
                                                     float* __restrict__ part) {
  __shared__ float lds[256];
  float s = 0.f;
  for (int i = blockIdx.x * 256 + threadIdx.x; i < n; i += 256 * 256) s += g[i] * g[i];
  lds[threadIdx.x] = s; __syncthreads();
  for (int off = 128; off > 0; off >>= 1) {
    if (threadIdx.x < off) lds[threadIdx.x] += lds[threadIdx.x + off];
    __syncthreads();
  }
  if (threadIdx.x == 0) part[blockIdx.x] = lds[0];
}

__global__ __launch_bounds__(256) void k_gg2_final(const float* __restrict__ part,
                                                   float* __restrict__ out) {
  __shared__ float lds[256];
  lds[threadIdx.x] = part[threadIdx.x];
  __syncthreads();
  for (int off = 128; off > 0; off >>= 1) {
    if (threadIdx.x < off) lds[threadIdx.x] += lds[threadIdx.x + off];
    __syncthreads();
  }
  if (threadIdx.x == 0) *out = lds[0];
}

// t[j] = whA[Pi[j] & 1023] * GG[j]
__global__ __launch_bounds__(256) void k_build_t(const float* __restrict__ whA,
                                                 const int* __restrict__ Pi,
                                                 const float* __restrict__ GG,
                                                 float* __restrict__ t, int n) {
  for (int i = blockIdx.x * 256 + threadIdx.x; i < n; i += gridDim.x * 256)
    t[i] = whA[Pi[i] & (DINT - 1)] * GG[i];
}

// FWHT over the 10 high bits (view t as [1024 rows][8192 cols]); 8 cols/block in LDS
__global__ __launch_bounds__(512) void k_fwht_hi(float* __restrict__ t) {
  __shared__ float lds[8192];          // 1024 rows x 8 cols, 32 KB
  const int tid = threadIdx.x;
  const size_t colBase = (size_t)blockIdx.x * 8;
  for (int e = tid; e < 8192; e += 512) {
    const int r = e >> 3, c = e & 7;
    lds[e] = t[(size_t)r * 8192 + colBase + c];
  }
  __syncthreads();
  for (int lev = 0; lev < 10; ++lev) {
    const int s = 1 << lev;
    for (int p = tid; p < 4096; p += 512) {
      const int c = p & 7, b = p >> 3;
      const int i = ((b >> lev) << (lev + 1)) | (b & (s - 1));
      const int j = i + s;
      const float a0 = lds[i * 8 + c], a1 = lds[j * 8 + c];
      lds[i * 8 + c] = a0 + a1;
      lds[j * 8 + c] = a0 - a1;
    }
    __syncthreads();
  }
  for (int e = tid; e < 8192; e += 512) {
    const int r = e >> 3, c = e & 7;
    t[(size_t)r * 8192 + colBase + c] = lds[e];
  }
}

// FWHT over the 13 low bits + fused epilogue: outb = bf16(W0 + val * rsqrt(LL*sumGG2))
__global__ __launch_bounds__(512) void k_fwht_lo_epi(const float* __restrict__ t,
                                                     const float* __restrict__ W0,
                                                     unsigned short* __restrict__ outb,
                                                     const float* __restrict__ sumptr,
                                                     float LLf) {
  __shared__ float lds[8192];          // 32 KB
  const int tid = threadIdx.x;
  const size_t base = (size_t)blockIdx.x * 8192;
  for (int e = tid; e < 8192; e += 512) lds[e] = t[base + e];
  __syncthreads();
  for (int lev = 0; lev < 13; ++lev) {
    const int s = 1 << lev;
    for (int p = tid; p < 4096; p += 512) {
      const int i = ((p >> lev) << (lev + 1)) | (p & (s - 1));
      const int j = i + s;
      const float a0 = lds[i], a1 = lds[j];
      lds[i] = a0 + a1; lds[j] = a0 - a1;
    }
    __syncthreads();
  }
  const float scale = rsqrtf(LLf * sumptr[0]);
  for (int e = tid; e < 8192; e += 512)
    outb[base + e] = f2bf(W0[base + e] + lds[e] * scale);
}

// Full small fastfood for biases (LL <= 4096 fits in LDS), fp32 output
__global__ __launch_bounds__(512) void k_bias_ff(const float* __restrict__ whA,
                                                 const int* __restrict__ Pi,
                                                 const float* __restrict__ GG,
                                                 const float* __restrict__ b0,
                                                 float* __restrict__ bout,
                                                 const float* __restrict__ sumptr,
                                                 int LL, int levels) {
  __shared__ float lds[4096];
  const int tid = threadIdx.x;
  for (int j = tid; j < LL; j += 512) lds[j] = whA[Pi[j] & (DINT - 1)] * GG[j];
  __syncthreads();
  const int half = LL >> 1;
  for (int lev = 0; lev < levels; ++lev) {
    const int s = 1 << lev;
    for (int p = tid; p < half; p += 512) {
      const int i = ((p >> lev) << (lev + 1)) | (p & (s - 1));
      const int j = i + s;
      const float a0 = lds[i], a1 = lds[j];
      lds[i] = a0 + a1; lds[j] = a0 - a1;
    }
    __syncthreads();
  }
  const float scale = rsqrtf((float)LL * sumptr[0]);
  for (int j = tid; j < LL; j += 512) bout[j] = b0[j] + lds[j] * scale;
}

__global__ __launch_bounds__(256) void k_cvt_bf16(const float* __restrict__ in,
                                                  unsigned short* __restrict__ out, int n) {
  for (int i = blockIdx.x * 256 + threadIdx.x; i < n; i += gridDim.x * 256)
    out[i] = f2bf(in[i]);
}

// ---------------- bf16 WMMA GEMM (NT: C[m,n] = sum_k A[m,k]*B[n,k]) ----------------

// A fragment (16x32, 16-bit): lanes 0-15 hold M=lane,K={0..7,16..23}; lanes 16-31 M=lane-16,K={8..15,24..31}
static __device__ inline v16bf load_fragA(const unsigned short* base, int ld,
                                          int row0, int k0, int lane) {
  const int m  = lane & 15;
  const int ka = (lane < 16) ? 0 : 8;
  const unsigned short* p = base + (size_t)(row0 + m) * ld + k0 + ka;
  FragU f;
  f.u[0] = *reinterpret_cast<const U128*>(p);        // K ka..ka+7
  f.u[1] = *reinterpret_cast<const U128*>(p + 16);   // K ka+16..ka+23
  return f.v;
}

// B fragment (32x16): lanes 0-15 hold N=lane,K={0..7,16..23}; lanes 16-31 N=lane-16,K={8..15,24..31}
static __device__ inline v16bf load_fragB(const unsigned short* base, int ld,
                                          int col0, int k0, int lane) {
  const int n  = lane & 15;
  const int ka = (lane < 16) ? 0 : 8;
  const unsigned short* p = base + (size_t)(col0 + n) * ld + k0 + ka;
  FragU f;
  f.u[0] = *reinterpret_cast<const U128*>(p);
  f.u[1] = *reinterpret_cast<const U128*>(p + 16);
  return f.v;
}

__global__ __launch_bounds__(256)
void k_gemm_bf16(const unsigned short* __restrict__ A,   // M x K row-major
                 const unsigned short* __restrict__ Bm,  // N x K row-major
                 const float* __restrict__ bias,
                 unsigned short* __restrict__ outB,      // mode 1: bf16 relu out
                 float* __restrict__ outF,               // mode 0: f32 out
                 int M, int N, int K, int fuse_relu_bf16) {
  const int lane = threadIdx.x & 31;
  const int wid  = threadIdx.x >> 5;                 // 8 waves (wave32)
  const int mBase = blockIdx.y * 64  + (wid & 1) * 32;
  const int nBase = blockIdx.x * 128 + (wid >> 1) * 32;
  (void)M;

  v8f acc00 = {}, acc01 = {}, acc10 = {}, acc11 = {};
  for (int k0 = 0; k0 < K; k0 += 32) {
    const v16bf a0 = load_fragA(A, K, mBase,      k0, lane);
    const v16bf a1 = load_fragA(A, K, mBase + 16, k0, lane);
    const v16bf b0 = load_fragB(Bm, K, nBase,      k0, lane);
    const v16bf b1 = load_fragB(Bm, K, nBase + 16, k0, lane);
    acc00 = __builtin_amdgcn_wmma_f32_16x16x32_bf16(false, a0, false, b0, (short)0, acc00, false, false);
    acc01 = __builtin_amdgcn_wmma_f32_16x16x32_bf16(false, a0, false, b1, (short)0, acc01, false, false);
    acc10 = __builtin_amdgcn_wmma_f32_16x16x32_bf16(false, a1, false, b0, (short)0, acc10, false, false);
    acc11 = __builtin_amdgcn_wmma_f32_16x16x32_bf16(false, a1, false, b1, (short)0, acc11, false, false);
  }

  // C/D layout: VGPR g, lanes<16 -> M=g, lanes>=16 -> M=g+8; N = lane&15
  const int rowOff = (lane < 16) ? 0 : 8;
  const int col = lane & 15;
  v8f accs[2][2] = {{acc00, acc01}, {acc10, acc11}};
  for (int i = 0; i < 2; ++i)
    for (int j = 0; j < 2; ++j) {
      const int n = nBase + j * 16 + col;
      const float bv = bias[n];
      for (int g = 0; g < 8; ++g) {
        const int m = mBase + i * 16 + rowOff + g;
        float v = accs[i][j][g] + bv;
        if (fuse_relu_bf16) {
          v = fmaxf(v, 0.0f);
          outB[(size_t)m * N + n] = f2bf(v);
        } else {
          outF[(size_t)m * N + n] = v;
        }
      }
    }
}

// ---------------- host launch ----------------

extern "C" void kernel_launch(void* const* d_in, const int* in_sizes, int n_in,
                              void* d_out, int out_size, void* d_ws, size_t ws_size,
                              hipStream_t stream) {
  (void)in_sizes; (void)n_in; (void)out_size; (void)ws_size;
  const float* x     = (const float*)d_in[0];
  const float* V     = (const float*)d_in[1];
  const float* W1_0  = (const float*)d_in[2];
  const float* b1_0  = (const float*)d_in[3];
  const float* W2_0  = (const float*)d_in[4];
  const float* b2_0  = (const float*)d_in[5];
  const float* BB_W1 = (const float*)d_in[6];
  const int*   Pi_W1 = (const int*)d_in[7];
  const float* GG_W1 = (const float*)d_in[8];
  const float* BB_b1 = (const float*)d_in[9];
  const int*   Pi_b1 = (const int*)d_in[10];
  const float* GG_b1 = (const float*)d_in[11];
  const float* BB_W2 = (const float*)d_in[12];
  const int*   Pi_W2 = (const int*)d_in[13];
  const float* GG_W2 = (const float*)d_in[14];
  const float* BB_b2 = (const float*)d_in[15];
  const int*   Pi_b2 = (const int*)d_in[16];
  const float* GG_b2 = (const float*)d_in[17];

  float* ws   = (float*)d_ws;
  float* sums = ws;                 // 4 scalars (padded to 16)
  float* part = ws + 16;            // 256 partials
  float* whA  = ws + 512;           // 4 x 1024
  float* t    = ws + 512 + 4096;    // 2^23 floats, shared by W1/W2 stages
  float* b1b  = t + LLW;            // 4096
  float* b2b  = b1b + 4096;         // 2048
  unsigned short* xb  = (unsigned short*)(b2b + 2048);          // 1024x2048 bf16
  unsigned short* w1b = xb  + (size_t)Bsz * DIN;                // 4096x2048 bf16
  unsigned short* w2b = w1b + (size_t)DH * DIN;                 // 2048x4096 bf16
  unsigned short* hb  = w2b + (size_t)DOUT * DH;                // 1024x4096 bf16

  // Collapsed first FWHTs (1024-point each)
  k_whA<<<1, 512, 0, stream>>>(V, BB_W1, whA + 0);
  k_whA<<<1, 512, 0, stream>>>(V, BB_b1, whA + 1024);
  k_whA<<<1, 512, 0, stream>>>(V, BB_W2, whA + 2048);
  k_whA<<<1, 512, 0, stream>>>(V, BB_b2, whA + 3072);

  // Deterministic two-stage sum(GG^2) reductions
  k_gg2_partial<<<256, 256, 0, stream>>>(GG_W1, LLW,  part);
  k_gg2_final  <<<1,   256, 0, stream>>>(part, sums + 0);
  k_gg2_partial<<<256, 256, 0, stream>>>(GG_b1, DH,   part);
  k_gg2_final  <<<1,   256, 0, stream>>>(part, sums + 1);
  k_gg2_partial<<<256, 256, 0, stream>>>(GG_W2, LLW,  part);
  k_gg2_final  <<<1,   256, 0, stream>>>(part, sums + 2);
  k_gg2_partial<<<256, 256, 0, stream>>>(GG_b2, DOUT, part);
  k_gg2_final  <<<1,   256, 0, stream>>>(part, sums + 3);

  // Biases (fully in-LDS fastfood)
  k_bias_ff<<<1, 512, 0, stream>>>(whA + 1024, Pi_b1, GG_b1, b1_0, b1b, sums + 1, DH,   12);
  k_bias_ff<<<1, 512, 0, stream>>>(whA + 3072, Pi_b2, GG_b2, b2_0, b2b, sums + 3, DOUT, 11);

  // x -> bf16
  k_cvt_bf16<<<1024, 256, 0, stream>>>(x, xb, Bsz * DIN);

  // W1 reconstruction: permute/gather -> high-bit FWHT -> low-bit FWHT + fused W0+scale -> bf16
  k_build_t    <<<2048, 256, 0, stream>>>(whA + 0, Pi_W1, GG_W1, t, LLW);
  k_fwht_hi    <<<1024, 512, 0, stream>>>(t);
  k_fwht_lo_epi<<<1024, 512, 0, stream>>>(t, W1_0, w1b, sums + 0, (float)LLW);

  // GEMM1: h = relu(x @ W1^T + b1), bf16 out
  k_gemm_bf16<<<dim3(DH / 128, Bsz / 64), 256, 0, stream>>>(
      xb, w1b, b1b, hb, nullptr, Bsz, DH, DIN, 1);

  // W2 reconstruction (reuses t)
  k_build_t    <<<2048, 256, 0, stream>>>(whA + 2048, Pi_W2, GG_W2, t, LLW);
  k_fwht_hi    <<<1024, 512, 0, stream>>>(t);
  k_fwht_lo_epi<<<1024, 512, 0, stream>>>(t, W2_0, w2b, sums + 2, (float)LLW);

  // GEMM2: out = h @ W2^T + b2, f32 out
  k_gemm_bf16<<<dim3(DOUT / 128, Bsz / 64), 256, 0, stream>>>(
      hb, w2b, b2b, nullptr, (float*)d_out, Bsz, DOUT, DH, 0);
}
